// IntegratedAttention_13872744366403
// MI455X (gfx1250) — matrix operand order repaired and checked by reading, not compiled, not run on hardware
//
#include <hip/hip_runtime.h>
#include <stdint.h>

#define S_LEN 2048
#define DMODEL 1024
#define NHEAD 16
#define HDIM 64

typedef __attribute__((ext_vector_type(8)))  float    v8f;
typedef __attribute__((ext_vector_type(16))) __bf16   bf16x16;
typedef __attribute__((ext_vector_type(8)))  uint16_t u16x8;

struct U16x16 { u16x8 lo, hi; };

static __device__ __forceinline__ uint16_t f2bf(float f) {
  uint32_t u = __float_as_uint(f);
  u += 0x7FFFu + ((u >> 16) & 1u);          // round-to-nearest-even
  return (uint16_t)(u >> 16);
}

// Build a 16-element bf16 fragment from two 16-byte chunks.
static __device__ __forceinline__ bf16x16 frag2(const uint16_t* p0, const uint16_t* p1) {
  U16x16 t;
  t.lo = *reinterpret_cast<const u16x8*>(p0);
  t.hi = *reinterpret_cast<const u16x8*>(p1);
  return __builtin_bit_cast(bf16x16, t);
}

static __device__ __forceinline__ v8f wmma_bf16(bf16x16 a, bf16x16 b, v8f c) {
  return __builtin_amdgcn_wmma_f32_16x16x32_bf16(false, a, false, b, (short)0, c, false, false);
}

// ---------------------------------------------------------------- convert
__global__ void cvt_bf16_kernel(const float* __restrict__ src,
                                uint16_t* __restrict__ dst, int n) {
  int i = blockIdx.x * blockDim.x + threadIdx.x;
  if (i < n) dst[i] = f2bf(src[i]);
}

// ---------------------------------------------------------------- QKV GEMM
// out[m,n] = sum_k x[m,k] * w[n,k] (+ bias[n]).  blockIdx.z: 0=Q 1=K 2=V.
// Wave tile 64x32 (4 M-tiles x 2 N-tiles = 8 WMMA per K-step).
// Q,K stored [b,h,s,d]; V stored transposed [b,h,d,s].
__global__ void qkv_kernel(const uint16_t* __restrict__ xb,
                           const uint16_t* __restrict__ wqb,
                           const uint16_t* __restrict__ wkb,
                           const uint16_t* __restrict__ wvb,
                           const float* __restrict__ bq,
                           const float* __restrict__ bv,
                           uint16_t* __restrict__ Qb,
                           uint16_t* __restrict__ Kb,
                           uint16_t* __restrict__ Vtb) {
  const int lane = threadIdx.x & 31;
  const int wave = threadIdx.x >> 5;
  const int lh   = lane >> 4;      // lane half
  const int r    = lane & 15;
  const int mode = blockIdx.z;
  const uint16_t* wb   = (mode == 0) ? wqb : (mode == 1) ? wkb : wvb;
  const float*    bias = (mode == 0) ? bq  : (mode == 2) ? bv  : nullptr;

  const int m0 = blockIdx.y * 128 + (wave >> 1) * 64;
  const int n0 = blockIdx.x * 64  + (wave & 1) * 32;

  v8f c[4][2] = {};
  const uint16_t* ab[4];
  #pragma unroll
  for (int i = 0; i < 4; ++i) ab[i] = xb + (size_t)(m0 + i * 16 + r) * DMODEL;
  const uint16_t* bb[2];
  #pragma unroll
  for (int j = 0; j < 2; ++j) bb[j] = wb + (size_t)(n0 + j * 16 + r) * DMODEL;

  for (int k0 = 0; k0 < DMODEL; k0 += 32) {
    const int ka = k0 + lh * 8;      // A operand: chunks at ka, ka+16
    const int kb = k0 + lh * 8;      // B operand: chunks at kb, kb+16
    // Unconditional speculative prefetch (WGP scope request): runs past the
    // tile end by <=512B; invalid translations are silently dropped.
    __builtin_prefetch(ab[0] + ka + 256, 0, 3);
    __builtin_prefetch(bb[0] + kb + 256, 0, 3);
    bf16x16 a[4], b[2];
    #pragma unroll
    for (int i = 0; i < 4; ++i) a[i] = frag2(ab[i] + ka, ab[i] + ka + 16);
    #pragma unroll
    for (int j = 0; j < 2; ++j) b[j] = frag2(bb[j] + kb, bb[j] + kb + 16);
    #pragma unroll
    for (int i = 0; i < 4; ++i)
      #pragma unroll
      for (int j = 0; j < 2; ++j)
        c[i][j] = wmma_bf16(a[i], b[j], c[i][j]);
  }

  #pragma unroll
  for (int i = 0; i < 4; ++i) {
    #pragma unroll
    for (int j = 0; j < 2; ++j) {
      const int mt = m0 + i * 16;
      const int nt = n0 + j * 16;
      #pragma unroll
      for (int r8 = 0; r8 < 8; ++r8) {
        const int m = mt + r8 + lh * 8;
        const int n = nt + r;
        float v = c[i][j][r8];
        if (bias) v += bias[n];
        const int b_ = m >> 11, s = m & 2047;
        const int h  = n >> 6,  d = n & 63;
        const uint16_t bf = f2bf(v);
        if (mode == 2) {
          Vtb[((size_t)(b_ * NHEAD + h) * HDIM + d) * S_LEN + s] = bf;
        } else {
          uint16_t* dst = (mode == 0) ? Qb : Kb;
          dst[((size_t)(b_ * NHEAD + h) * S_LEN + s) * HDIM + d] = bf;
        }
      }
    }
  }
}

// ------------------------------------------------ online-softmax path step
static __device__ __forceinline__ void path_update(
    const v8f sc[4], const float rmax[8], float fscale,
    float m[8], float l[8], v8f acc[4],
    uint16_t* plds, int lh, int r,
    const uint16_t* vh, int kt0) {
  float p[4][8];
  #pragma unroll
  for (int r8 = 0; r8 < 8; ++r8) {
    const float mo = m[r8];
    const float mn = fmaxf(mo, fscale * rmax[r8]);
    const float alpha = __expf(mo - mn);
    float rs = 0.f;
    #pragma unroll
    for (int nt = 0; nt < 4; ++nt) {
      const float e = __expf(fscale * sc[nt][r8] - mn);
      p[nt][r8] = e;
      rs += e;
    }
    rs += __shfl_xor(rs, 1, 32);
    rs += __shfl_xor(rs, 2, 32);
    rs += __shfl_xor(rs, 4, 32);
    rs += __shfl_xor(rs, 8, 32);
    l[r8] = l[r8] * alpha + rs;
    m[r8] = mn;
    #pragma unroll
    for (int t = 0; t < 4; ++t) acc[t][r8] *= alpha;
  }
  // C-layout -> row-major 16x64 bf16 in LDS, then reload as A-fragments.
  #pragma unroll
  for (int nt = 0; nt < 4; ++nt) {
    #pragma unroll
    for (int r8 = 0; r8 < 8; ++r8) {
      plds[(r8 + lh * 8) * 64 + nt * 16 + r] = f2bf(p[nt][r8]);
    }
  }
  const uint16_t* prow = plds + r * 64;
  bf16x16 pa0 = frag2(prow + lh * 8,      prow + lh * 8 + 16);       // keys 0..31
  bf16x16 pa1 = frag2(prow + 32 + lh * 8, prow + 32 + lh * 8 + 16);  // keys 32..63
  #pragma unroll
  for (int t = 0; t < 4; ++t) {   // d-tiles
    const uint16_t* vrow = vh + (size_t)(t * 16 + r) * S_LEN + kt0;
    bf16x16 bv0 = frag2(vrow + lh * 8,      vrow + lh * 8 + 16);
    bf16x16 bv1 = frag2(vrow + 32 + lh * 8, vrow + 32 + lh * 8 + 16);
    acc[t] = wmma_bf16(pa0, bv0, acc[t]);
    acc[t] = wmma_bf16(pa1, bv1, acc[t]);
  }
}

// ---------------------------------------------------------------- attention
// grid (S/64, B*H), block 128: each wave owns 16 query rows of one head.
__global__ void attn_kernel(const uint16_t* __restrict__ Qb,
                            const uint16_t* __restrict__ Kb,
                            const uint16_t* __restrict__ Vtb,
                            const float* __restrict__ span_scale,
                            float* __restrict__ out,          // span path, f32 [B,S,D]
                            uint16_t* __restrict__ Omain) {   // main path, bf16 [B*S, D]
  __shared__ __align__(16) uint16_t Plds[4][16 * 64];
  const int lane = threadIdx.x & 31;
  const int wave = threadIdx.x >> 5;
  const int lh   = lane >> 4;
  const int r    = lane & 15;
  const int bh   = blockIdx.y;
  const int b    = bh >> 4, h = bh & 15;
  const int m0   = blockIdx.x * 64 + wave * 16;

  const uint16_t* qh = Qb  + (size_t)bh * S_LEN * HDIM;
  const uint16_t* kh = Kb  + (size_t)bh * S_LEN * HDIM;
  const uint16_t* vh = Vtb + (size_t)bh * HDIM * S_LEN;

  const float temperature = 1.0f + 0.01f * (1.0f - span_scale[0]);
  const float f_span = 0.35355339059327373f / temperature;  // hd^-0.25 / T
  const float f_main = 1.0f / 64.0f;                        // hd^-0.5 / sqrt(hd)

  const uint16_t* qrow = qh + (size_t)(m0 + r) * HDIM;
  bf16x16 aq0 = frag2(qrow + lh * 8,      qrow + lh * 8 + 16);
  bf16x16 aq1 = frag2(qrow + 32 + lh * 8, qrow + 32 + lh * 8 + 16);

  v8f accS[4] = {}, accM[4] = {};
  float mS[8], lS[8], mM[8], lM[8];
  #pragma unroll
  for (int i = 0; i < 8; ++i) { mS[i] = -1e30f; lS[i] = 0.f; mM[i] = -1e30f; lM[i] = 0.f; }

  for (int kt0 = 0; kt0 < S_LEN; kt0 += 64) {
    v8f sc[4] = {};
    #pragma unroll
    for (int nt = 0; nt < 4; ++nt) {
      const uint16_t* krow = kh + (size_t)(kt0 + nt * 16 + r) * HDIM;
      bf16x16 bk0 = frag2(krow + lh * 8,      krow + lh * 8 + 16);
      bf16x16 bk1 = frag2(krow + 32 + lh * 8, krow + 32 + lh * 8 + 16);
      sc[nt] = wmma_bf16(aq0, bk0, sc[nt]);
      sc[nt] = wmma_bf16(aq1, bk1, sc[nt]);
    }
    float rmax[8];
    #pragma unroll
    for (int r8 = 0; r8 < 8; ++r8) {
      float v = fmaxf(fmaxf(sc[0][r8], sc[1][r8]), fmaxf(sc[2][r8], sc[3][r8]));
      v = fmaxf(v, __shfl_xor(v, 1, 32));
      v = fmaxf(v, __shfl_xor(v, 2, 32));
      v = fmaxf(v, __shfl_xor(v, 4, 32));
      v = fmaxf(v, __shfl_xor(v, 8, 32));
      rmax[r8] = v;
    }
    path_update(sc, rmax, f_span, mS, lS, accS, &Plds[wave][0], lh, r, vh, kt0);
    path_update(sc, rmax, f_main, mM, lM, accM, &Plds[wave][0], lh, r, vh, kt0);
  }

  #pragma unroll
  for (int r8 = 0; r8 < 8; ++r8) {
    const float invS = 1.0f / lS[r8];
    const float invM = 1.0f / lM[r8];
    const int   srow = m0 + r8 + lh * 8;
    #pragma unroll
    for (int t = 0; t < 4; ++t) {
      const int d = t * 16 + r;
      const size_t oidx = (size_t)(b * S_LEN + srow) * DMODEL + h * HDIM + d;
      out[oidx]   = accS[t][r8] * invS;
      Omain[oidx] = f2bf(accM[t][r8] * invM);
    }
  }
}

// ------------------------------------------------------ output projection
// out[m,n] += sum_k A[m,k]*wo[n,k] + bo[n]   (out already holds span_out)
// Wave tile 64x32, same as qkv_kernel.
__global__ void proj_kernel(const uint16_t* __restrict__ Ab,
                            const uint16_t* __restrict__ wb,
                            const float* __restrict__ bias,
                            float* __restrict__ out) {
  const int lane = threadIdx.x & 31;
  const int wave = threadIdx.x >> 5;
  const int lh   = lane >> 4;
  const int r    = lane & 15;
  const int m0 = blockIdx.y * 128 + (wave >> 1) * 64;
  const int n0 = blockIdx.x * 64  + (wave & 1) * 32;

  v8f c[4][2] = {};
  const uint16_t* ab[4];
  #pragma unroll
  for (int i = 0; i < 4; ++i) ab[i] = Ab + (size_t)(m0 + i * 16 + r) * DMODEL;
  const uint16_t* bb[2];
  #pragma unroll
  for (int j = 0; j < 2; ++j) bb[j] = wb + (size_t)(n0 + j * 16 + r) * DMODEL;

  for (int k0 = 0; k0 < DMODEL; k0 += 32) {
    const int ka = k0 + lh * 8;
    const int kb = k0 + lh * 8;
    __builtin_prefetch(ab[0] + ka + 256, 0, 3);
    __builtin_prefetch(bb[0] + kb + 256, 0, 3);
    bf16x16 a[4], b[2];
    #pragma unroll
    for (int i = 0; i < 4; ++i) a[i] = frag2(ab[i] + ka, ab[i] + ka + 16);
    #pragma unroll
    for (int j = 0; j < 2; ++j) b[j] = frag2(bb[j] + kb, bb[j] + kb + 16);
    #pragma unroll
    for (int i = 0; i < 4; ++i)
      #pragma unroll
      for (int j = 0; j < 2; ++j)
        c[i][j] = wmma_bf16(a[i], b[j], c[i][j]);
  }

  #pragma unroll
  for (int i = 0; i < 4; ++i) {
    #pragma unroll
    for (int j = 0; j < 2; ++j) {
      const int mt = m0 + i * 16;
      const int nt = n0 + j * 16;
      #pragma unroll
      for (int r8 = 0; r8 < 8; ++r8) {
        const int m = mt + r8 + lh * 8;
        const int n = nt + r;
        const size_t idx = (size_t)m * DMODEL + n;
        out[idx] += c[i][j][r8] + bias[n];
      }
    }
  }
}

// ---------------------------------------------------------------- launcher
extern "C" void kernel_launch(void* const* d_in, const int* in_sizes, int n_in,
                              void* d_out, int out_size, void* d_ws, size_t ws_size,
                              hipStream_t stream) {
  (void)in_sizes; (void)n_in; (void)out_size; (void)ws_size;
  const float* x  = (const float*)d_in[0];
  const float* wq = (const float*)d_in[1];
  const float* bq = (const float*)d_in[2];
  const float* wk = (const float*)d_in[3];
  const float* wv = (const float*)d_in[4];
  const float* bv = (const float*)d_in[5];
  const float* wo = (const float*)d_in[6];
  const float* bo = (const float*)d_in[7];
  const float* span_scale = (const float*)d_in[8];
  float* out = (float*)d_out;

  char* ws = (char*)d_ws;
  uint16_t* xb    = (uint16_t*)(ws);               //  8 MiB: x bf16 [4096,1024]
  uint16_t* wqb   = (uint16_t*)(ws +  8388608);    //  2 MiB
  uint16_t* wkb   = (uint16_t*)(ws + 10485760);    //  2 MiB
  uint16_t* wvb   = (uint16_t*)(ws + 12582912);    //  2 MiB
  uint16_t* wob   = (uint16_t*)(ws + 14680064);    //  2 MiB
  uint16_t* Qb    = (uint16_t*)(ws + 16777216);    //  8 MiB [b,h,s,d]
  uint16_t* Kb    = (uint16_t*)(ws + 25165824);    //  8 MiB [b,h,s,d]
  uint16_t* Vtb   = (uint16_t*)(ws + 33554432);    //  8 MiB [b,h,d,s]
  uint16_t* Omain = (uint16_t*)(ws + 41943040);    //  8 MiB [b*s, d]

  const int nx = 4096 * 1024, nw = 1024 * 1024;
  cvt_bf16_kernel<<<(nx + 255) / 256, 256, 0, stream>>>(x,  xb,  nx);
  cvt_bf16_kernel<<<(nw + 255) / 256, 256, 0, stream>>>(wq, wqb, nw);
  cvt_bf16_kernel<<<(nw + 255) / 256, 256, 0, stream>>>(wk, wkb, nw);
  cvt_bf16_kernel<<<(nw + 255) / 256, 256, 0, stream>>>(wv, wvb, nw);
  cvt_bf16_kernel<<<(nw + 255) / 256, 256, 0, stream>>>(wo, wob, nw);

  qkv_kernel<<<dim3(16, 32, 3), 128, 0, stream>>>(xb, wqb, wkb, wvb, bq, bv,
                                                  Qb, Kb, Vtb);
  attn_kernel<<<dim3(32, 32), 128, 0, stream>>>(Qb, Kb, Vtb, span_scale,
                                                out, Omain);
  proj_kernel<<<dim3(16, 32), 128, 0, stream>>>(Omain, wob, bo, out);
}
